// OpenAIPrivacyFilterEncoderLayer_34471407518015
// MI455X (gfx1250) — matrix operand order repaired and checked by reading, not compiled, not run on hardware
//
#include <hip/hip_runtime.h>
#include <hip/hip_bf16.h>

typedef __attribute__((ext_vector_type(16))) __bf16 v16bf;
typedef __attribute__((ext_vector_type(8)))  float  v8f;

#define TNUM 4096
#define SEQ  2048
#define HID  1024
#define LSTR 56          // LDS stride (elems) for GEMM tiles: 112B, 16B-aligned, conflict-friendly
#define ASTR 40          // LDS stride for attention P/V tiles

__device__ __forceinline__ float bf2f(__bf16 b) {
  unsigned short s = __builtin_bit_cast(unsigned short, b);
  unsigned u = ((unsigned)s) << 16;
  return __builtin_bit_cast(float, u);
}
__device__ __forceinline__ __bf16 f2bf(float f) {
  unsigned u = __builtin_bit_cast(unsigned, f);
  unsigned r = u + 0x7FFFu + ((u >> 16) & 1u);   // round-to-nearest-even
  unsigned short s = (unsigned short)(r >> 16);
  return __builtin_bit_cast(__bf16, s);
}

union FragU { uint4 q[2]; v16bf v; };
// A-frag style: 8 elems at p, 8 elems at p+16 (K and K+16 halves)
__device__ __forceinline__ v16bf ld_split(const __bf16* p) {
  FragU f;
  f.q[0] = *reinterpret_cast<const uint4*>(p);
  f.q[1] = *reinterpret_cast<const uint4*>(p + 16);
  return f.v;
}
// B-frag style: 16 contiguous elems
__device__ __forceinline__ v16bf ld_cont(const __bf16* p) {
  FragU f;
  f.q[0] = *reinterpret_cast<const uint4*>(p);
  f.q[1] = *reinterpret_cast<const uint4*>(p + 8);
  return f.v;
}
__device__ __forceinline__ v8f vzero8() {
  v8f z;
#pragma unroll
  for (int i = 0; i < 8; ++i) z[i] = 0.f;
  return z;
}

// CDNA5 async global->LDS copy, 16 bytes per lane (ASYNCcnt-tracked DMA path).
// Low 32 bits of a generic shared-aperture address are the LDS byte offset.
__device__ __forceinline__ void async_copy_b128(const void* gptr, void* lptr) {
  unsigned lds = (unsigned)(size_t)lptr;
  unsigned long long ga = (unsigned long long)(size_t)gptr;
  asm volatile("global_load_async_to_lds_b128 %0, %1, off"
               :: "v"(lds), "v"(ga) : "memory");
}
__device__ __forceinline__ void wait_async0() {
  asm volatile("s_wait_asynccnt 0" ::: "memory");
}

// ---------------- RMSNorm: fp32 [T,1024] -> bf16 [T,1024] ----------------
__global__ __launch_bounds__(256) void rmsnorm_kernel(const float* __restrict__ x,
                                                      const float* __restrict__ scale,
                                                      __bf16* __restrict__ out) {
  const int t = blockIdx.x;
  const float* xr = x + (size_t)t * HID;
  __shared__ float red[8];
  float ss = 0.f;
  for (int i = threadIdx.x; i < HID; i += 256) { float v = xr[i]; ss += v * v; }
#pragma unroll
  for (int off = 16; off > 0; off >>= 1) ss += __shfl_xor(ss, off, 32);
  const int wid = threadIdx.x >> 5, lane = threadIdx.x & 31;
  if (lane == 0) red[wid] = ss;
  __syncthreads();
  float tot = 0.f;
#pragma unroll
  for (int i = 0; i < 8; ++i) tot += red[i];
  const float inv = rsqrtf(tot / (float)HID + 1e-5f);
  __bf16* o = out + (size_t)t * HID;
  for (int i = threadIdx.x; i < HID; i += 256) o[i] = f2bf(xr[i] * inv * scale[i]);
}

// ---------------- Generic WMMA GEMM: C = A(bf16)[M,K] * B(f32)[K,N] ----------------
// epi 0: out[m][ocol+n] = acc + bias[n]                (fp32 out, ldout)
// epi 1: out[m][n]      = resid[m][n] + acc + bias[n]  (attn residual)
// epi 2: out[m][n]     += scores[m*8+expert]*(acc + bias[n])  (MoE down accumulate)
__global__ __launch_bounds__(256) void gemm_kernel(const __bf16* __restrict__ A, int lda,
                                                   const float* __restrict__ B, int ldb,
                                                   const float* __restrict__ bias,
                                                   float* __restrict__ out, int ldout, int ocol,
                                                   const float* __restrict__ resid,
                                                   const float* __restrict__ scores, int expert,
                                                   int K, int epi) {
  __shared__ alignas(16) __bf16 As[128 * LSTR];
  __shared__ alignas(16) __bf16 Bt[128 * LSTR];   // transposed: Bt[n][k]
  const int tid = threadIdx.x;
  const int wid = tid >> 5, lane = tid & 31;
  const int wm = wid & 1, wn = wid >> 1;          // 2 x 4 wave grid; wave tile 64x32
  const int r = lane & 15, hi = lane >> 4;

  v8f acc[4][2];
#pragma unroll
  for (int i = 0; i < 4; ++i)
#pragma unroll
    for (int j = 0; j < 2; ++j) acc[i][j] = vzero8();

  const int m0 = blockIdx.x * 128;
  const int n0 = blockIdx.y * 128;

  for (int kb = 0; kb < K; kb += 32) {
    { // A tile 128x32 bf16: async DMA straight into LDS (2 x 16B per lane)
      const int row = tid >> 1, cs = (tid & 1) * 16;
      const __bf16* src = A + (size_t)(m0 + row) * lda + kb + cs;
      if (kb + 32 < K) __builtin_prefetch(src + 32, 0, 3);
      async_copy_b128(src,     &As[row * LSTR + cs]);
      async_copy_b128(src + 8, &As[row * LSTR + cs + 8]);
    }
    { // load B tile 32x128 f32 -> bf16, transposed into Bt[n][k]
      const int kk = tid >> 3, ncs = (tid & 7) * 16;
      const float* src = B + (size_t)(kb + kk) * ldb + n0 + ncs;
      if (kb + 32 < K) __builtin_prefetch(src + (size_t)32 * ldb, 0, 3);
      union { float4 f4[4]; float f[16]; } u;
      const float4* s4 = reinterpret_cast<const float4*>(src);
      u.f4[0] = s4[0]; u.f4[1] = s4[1]; u.f4[2] = s4[2]; u.f4[3] = s4[3];
#pragma unroll
      for (int i = 0; i < 16; ++i) Bt[(ncs + i) * LSTR + kk] = f2bf(u.f[i]);
    }
    wait_async0();
    __syncthreads();

    v16bf af[4], bfg[2];
#pragma unroll
    for (int i = 0; i < 4; ++i)
      af[i] = ld_cont(&As[(wm * 64 + i * 16 + r) * LSTR + hi * 16]);
#pragma unroll
    for (int j = 0; j < 2; ++j)
      bfg[j] = ld_cont(&Bt[(wn * 32 + j * 16 + r) * LSTR + hi * 16]);
#pragma unroll
    for (int i = 0; i < 4; ++i)
#pragma unroll
      for (int j = 0; j < 2; ++j)
        acc[i][j] = __builtin_amdgcn_wmma_f32_16x16x32_bf16(
            false, af[i], false, bfg[j], (short)0, acc[i][j], false, false);
    __syncthreads();
  }

#pragma unroll
  for (int i = 0; i < 4; ++i)
#pragma unroll
    for (int j = 0; j < 2; ++j)
#pragma unroll
      for (int v = 0; v < 8; ++v) {
        const int m = m0 + wm * 64 + i * 16 + hi * 8 + v;
        const int n = n0 + wn * 32 + j * 16 + r;
        const float val = acc[i][j][v];
        if (epi == 0) {
          out[(size_t)m * ldout + ocol + n] = val + bias[n];
        } else if (epi == 1) {
          out[(size_t)m * ldout + n] = resid[(size_t)m * ldout + n] + val + bias[n];
        } else {
          const float sc = scores[(size_t)m * 8 + expert];
          out[(size_t)m * ldout + n] += sc * (val + bias[n]);
        }
      }
}

// ---------------- RoPE + scale: qkv fp32 [T,1536] -> q/k/v bf16 ----------------
__global__ __launch_bounds__(256) void rope_kernel(const float* __restrict__ qkv,
                                                   __bf16* __restrict__ qb,
                                                   __bf16* __restrict__ kb,
                                                   __bf16* __restrict__ vb) {
  const int t = blockIdx.x;
  const int s = t & (SEQ - 1);
  const float* src = qkv + (size_t)t * 1536;
  const float lr = logf(150000.0f) / 32.0f;
  const float SC = 0.3535533906f; // 64^-0.25
  // q: 16 heads * 32 pairs
  for (int p = threadIdx.x; p < 512; p += 256) {
    const int h = p >> 5, d = p & 31;
    const float x1 = src[h * 64 + 2 * d], x2 = src[h * 64 + 2 * d + 1];
    const float fr = __expf(-(float)d * lr);
    float sn, cs; __sincosf((float)s * fr, &sn, &cs);
    qb[(size_t)t * HID + h * 64 + 2 * d]     = f2bf((x1 * cs - x2 * sn) * SC);
    qb[(size_t)t * HID + h * 64 + 2 * d + 1] = f2bf((x2 * cs + x1 * sn) * SC);
  }
  // k: 4 heads * 32 pairs
  for (int p = threadIdx.x; p < 128; p += 256) {
    const int h = p >> 5, d = p & 31;
    const float x1 = src[1024 + h * 64 + 2 * d], x2 = src[1024 + h * 64 + 2 * d + 1];
    const float fr = __expf(-(float)d * lr);
    float sn, cs; __sincosf((float)s * fr, &sn, &cs);
    kb[(size_t)t * 256 + h * 64 + 2 * d]     = f2bf((x1 * cs - x2 * sn) * SC);
    kb[(size_t)t * 256 + h * 64 + 2 * d + 1] = f2bf((x2 * cs + x1 * sn) * SC);
  }
  // v copy
  for (int i = threadIdx.x; i < 256; i += 256)
    vb[(size_t)t * 256 + i] = f2bf(src[1280 + i]);
}

// ---------------- Flash attention, sliding window 128 + sink, one wave/block ----------------
__global__ __launch_bounds__(32) void attn_kernel(const __bf16* __restrict__ qb,
                                                  const __bf16* __restrict__ kb,
                                                  const __bf16* __restrict__ vb,
                                                  const float* __restrict__ sinks,
                                                  __bf16* __restrict__ ob) {
  __shared__ alignas(16) __bf16 Ps[16 * ASTR];
  __shared__ alignas(16) __bf16 Vt[64 * ASTR];
  const int lane = threadIdx.x;
  const int r = lane & 15, hi = lane >> 4;
  const int q0 = blockIdx.x * 16;
  const int qh = blockIdx.y;
  const int b  = blockIdx.z;
  const int kvh = qh >> 2;

  // Q fragments (16x64 -> two 16x32 A-frags)
  const __bf16* qp = qb + ((size_t)(b * SEQ + q0 + r)) * HID + qh * 64;
  v16bf aq[2];
#pragma unroll
  for (int c = 0; c < 2; ++c) aq[c] = ld_cont(qp + c * 32 + hi * 16);

  v8f o[4];
#pragma unroll
  for (int n = 0; n < 4; ++n) o[n] = vzero8();
  float mx[8], ls[8];
#pragma unroll
  for (int v = 0; v < 8; ++v) { mx[v] = -1e30f; ls[v] = 0.f; }

  int jlo = q0 - 128; if (jlo < 0) jlo = 0; jlo &= ~31;
  int jhi = q0 + 144; if (jhi > SEQ) jhi = SEQ;

  for (int kc = jlo; kc < jhi; kc += 32) {
    // stage V chunk transposed: Vt[dim][key]
    {
      const __bf16* sv = vb + ((size_t)(b * SEQ + kc + lane)) * 256 + kvh * 64;
#pragma unroll
      for (int d8 = 0; d8 < 8; ++d8) {
        union { uint4 u; __bf16 h[8]; } w;
        w.u = reinterpret_cast<const uint4*>(sv)[d8];
#pragma unroll
        for (int j = 0; j < 8; ++j) Vt[(d8 * 8 + j) * ASTR + lane] = w.h[j];
      }
    }
    // S = Q K^T  (two 16-key n-tiles, K=64 in two chunks)
    v8f sf[2]; sf[0] = vzero8(); sf[1] = vzero8();
#pragma unroll
    for (int nt = 0; nt < 2; ++nt) {
      const __bf16* kp = kb + ((size_t)(b * SEQ + kc + nt * 16 + r)) * 256 + kvh * 64;
#pragma unroll
      for (int c = 0; c < 2; ++c) {
        v16bf bk = ld_cont(kp + c * 32 + hi * 16);
        sf[nt] = __builtin_amdgcn_wmma_f32_16x16x32_bf16(
            false, aq[c], false, bk, (short)0, sf[nt], false, false);
      }
    }
    // mask + streaming softmax update
#pragma unroll
    for (int v = 0; v < 8; ++v) {
      const int i = q0 + hi * 8 + v;
#pragma unroll
      for (int nt = 0; nt < 2; ++nt) {
        const int j = kc + nt * 16 + r;
        const int d = i - j;
        if (d < -128 || d > 128) sf[nt][v] = -1e9f;
      }
      float cm = fmaxf(sf[0][v], sf[1][v]);
#pragma unroll
      for (int off = 1; off < 16; off <<= 1) cm = fmaxf(cm, __shfl_xor(cm, off, 32));
      const float mn = fmaxf(mx[v], cm);
      const float rs = __expf(mx[v] - mn);
      mx[v] = mn;
      const float p0 = __expf(sf[0][v] - mn);
      const float p1 = __expf(sf[1][v] - mn);
      float psum = p0 + p1;
#pragma unroll
      for (int off = 1; off < 16; off <<= 1) psum += __shfl_xor(psum, off, 32);
      ls[v] = ls[v] * rs + psum;
#pragma unroll
      for (int n = 0; n < 4; ++n) o[n][v] *= rs;
      Ps[(hi * 8 + v) * ASTR + r]      = f2bf(p0);
      Ps[(hi * 8 + v) * ASTR + 16 + r] = f2bf(p1);
    }
    __syncthreads();
    // O += P V
    v16bf ap = ld_cont(&Ps[r * ASTR + hi * 16]);
#pragma unroll
    for (int n = 0; n < 4; ++n) {
      v16bf bv = ld_cont(&Vt[(n * 16 + r) * ASTR + hi * 16]);
      o[n] = __builtin_amdgcn_wmma_f32_16x16x32_bf16(
          false, ap, false, bv, (short)0, o[n], false, false);
    }
    __syncthreads();
  }
  // sink column
  const float sk = sinks[qh];
#pragma unroll
  for (int v = 0; v < 8; ++v) {
    const float mn = fmaxf(mx[v], sk);
    const float rs = __expf(mx[v] - mn);
    ls[v] = ls[v] * rs + __expf(sk - mn);
#pragma unroll
    for (int n = 0; n < 4; ++n) o[n][v] *= rs;
    mx[v] = mn;
  }
  // normalize + store
#pragma unroll
  for (int n = 0; n < 4; ++n)
#pragma unroll
    for (int v = 0; v < 8; ++v) {
      const int row = hi * 8 + v;
      const float val = o[n][v] / ls[v];
      ob[((size_t)(b * SEQ + q0 + row)) * HID + qh * 64 + n * 16 + r] = f2bf(val);
    }
}

// ---------------- Router: recompute rmsnorm in fp32, logits, top-4, softmax/4 ----------------
__global__ __launch_bounds__(256) void router_kernel(const float* __restrict__ h,
                                                     const float* __restrict__ nscale,
                                                     const float* __restrict__ rW,
                                                     const float* __restrict__ rb,
                                                     float* __restrict__ scores) {
  const int t = blockIdx.x;
  const float* xr = h + (size_t)t * HID;
  __shared__ float red[8];
  __shared__ float red2[64];
  float ss = 0.f;
  for (int i = threadIdx.x; i < HID; i += 256) { float v = xr[i]; ss += v * v; }
#pragma unroll
  for (int off = 16; off > 0; off >>= 1) ss += __shfl_xor(ss, off, 32);
  const int wid = threadIdx.x >> 5, lane = threadIdx.x & 31;
  if (lane == 0) red[wid] = ss;
  __syncthreads();
  float tot = 0.f;
#pragma unroll
  for (int i = 0; i < 8; ++i) tot += red[i];
  const float inv = rsqrtf(tot / (float)HID + 1e-5f);

  float pl[8];
#pragma unroll
  for (int e = 0; e < 8; ++e) pl[e] = 0.f;
  for (int i = threadIdx.x; i < HID; i += 256) {
    const float xn = xr[i] * inv * nscale[i];
#pragma unroll
    for (int e = 0; e < 8; ++e) pl[e] += xn * rW[i * 8 + e];
  }
#pragma unroll
  for (int e = 0; e < 8; ++e) {
#pragma unroll
    for (int off = 16; off > 0; off >>= 1) pl[e] += __shfl_xor(pl[e], off, 32);
    if (lane == 0) red2[wid * 8 + e] = pl[e];
  }
  __syncthreads();
  if (threadIdx.x == 0) {
    float lg[8];
#pragma unroll
    for (int e = 0; e < 8; ++e) {
      float s = rb[e];
#pragma unroll
      for (int w = 0; w < 8; ++w) s += red2[w * 8 + e];
      lg[e] = s;
    }
    int sel[4]; float wv[4];
#pragma unroll
    for (int k = 0; k < 4; ++k) {
      float best = -1e30f; int bi = 0;
#pragma unroll
      for (int e = 0; e < 8; ++e)
        if (lg[e] > best) { best = lg[e]; bi = e; }
      sel[k] = bi; wv[k] = best; lg[bi] = -1e30f;
    }
    const float m = wv[0];
    float sum = 0.f;
#pragma unroll
    for (int k = 0; k < 4; ++k) { wv[k] = __expf(wv[k] - m); sum += wv[k]; }
    float outv[8];
#pragma unroll
    for (int e = 0; e < 8; ++e) outv[e] = 0.f;
#pragma unroll
    for (int k = 0; k < 4; ++k) outv[sel[k]] = wv[k] / (sum * 4.0f);
#pragma unroll
    for (int e = 0; e < 8; ++e) scores[(size_t)t * 8 + e] = outv[e];
  }
}

// ---------------- Fused gate_up GEMM + clip + GLU -> gated bf16 [T,1024] ----------------
__global__ __launch_bounds__(256) void gateup_kernel(const __bf16* __restrict__ A,
                                                     const float* __restrict__ W,   // [1024][2048] expert base
                                                     const float* __restrict__ bias,// [2048] expert base
                                                     __bf16* __restrict__ outg) {
  __shared__ alignas(16) __bf16 As[128 * LSTR];
  __shared__ alignas(16) __bf16 Bg[64 * LSTR];
  __shared__ alignas(16) __bf16 Bu[64 * LSTR];
  const int tid = threadIdx.x;
  const int wid = tid >> 5, lane = tid & 31;
  const int wm = wid & 3, wn = wid >> 2;   // 4 x 2 wave grid; wave tile 32x32 (x2 weights)
  const int r = lane & 15, hi = lane >> 4;
  const int m0 = blockIdx.x * 128;
  const int n0 = blockIdx.y * 64;

  v8f cg[2][2], cu[2][2];
#pragma unroll
  for (int i = 0; i < 2; ++i)
#pragma unroll
    for (int j = 0; j < 2; ++j) { cg[i][j] = vzero8(); cu[i][j] = vzero8(); }

  for (int kb = 0; kb < HID; kb += 32) {
    { // A tile: async DMA into LDS
      const int row = tid >> 1, cs = (tid & 1) * 16;
      const __bf16* src = A + (size_t)(m0 + row) * HID + kb + cs;
      async_copy_b128(src,     &As[row * LSTR + cs]);
      async_copy_b128(src + 8, &As[row * LSTR + cs + 8]);
    }
    {
      const int sel = tid >> 7;               // 0 -> gate, 1 -> up
      const int tt = tid & 127;
      const int kk = tt >> 2, ncs = (tt & 3) * 16;
      const float* src = W + (size_t)(kb + kk) * 2048 + sel * 1024 + n0 + ncs;
      if (kb + 32 < HID) __builtin_prefetch(src + (size_t)32 * 2048, 0, 3);
      union { float4 f4[4]; float f[16]; } u;
      const float4* s4 = reinterpret_cast<const float4*>(src);
      u.f4[0] = s4[0]; u.f4[1] = s4[1]; u.f4[2] = s4[2]; u.f4[3] = s4[3];
      __bf16* dst = sel ? Bu : Bg;
#pragma unroll
      for (int i = 0; i < 16; ++i) dst[(ncs + i) * LSTR + kk] = f2bf(u.f[i]);
    }
    wait_async0();
    __syncthreads();

    v16bf af[2], bg[2], bu[2];
#pragma unroll
    for (int i = 0; i < 2; ++i)
      af[i] = ld_cont(&As[(wm * 32 + i * 16 + r) * LSTR + hi * 16]);
#pragma unroll
    for (int j = 0; j < 2; ++j) {
      bg[j] = ld_cont(&Bg[(wn * 32 + j * 16 + r) * LSTR + hi * 16]);
      bu[j] = ld_cont(&Bu[(wn * 32 + j * 16 + r) * LSTR + hi * 16]);
    }
#pragma unroll
    for (int i = 0; i < 2; ++i)
#pragma unroll
      for (int j = 0; j < 2; ++j) {
        cg[i][j] = __builtin_amdgcn_wmma_f32_16x16x32_bf16(
            false, af[i], false, bg[j], (short)0, cg[i][j], false, false);
        cu[i][j] = __builtin_amdgcn_wmma_f32_16x16x32_bf16(
            false, af[i], false, bu[j], (short)0, cu[i][j], false, false);
      }
    __syncthreads();
  }

#pragma unroll
  for (int i = 0; i < 2; ++i)
#pragma unroll
    for (int j = 0; j < 2; ++j)
#pragma unroll
      for (int v = 0; v < 8; ++v) {
        const int m = m0 + wm * 32 + i * 16 + hi * 8 + v;
        const int ng = n0 + wn * 32 + j * 16 + r;
        float g = cg[i][j][v] + bias[ng];
        float u = cu[i][j][v] + bias[1024 + ng];
        g = fminf(fmaxf(g, -1e9f), 7.0f);
        u = fminf(fmaxf(u, -7.0f), 7.0f);
        const float glu = g / (1.0f + __expf(-1.702f * g));
        outg[(size_t)m * HID + ng] = f2bf((u + 1.0f) * glu);
      }
}

// ------------------------------- launch -------------------------------
extern "C" void kernel_launch(void* const* d_in, const int* in_sizes, int n_in,
                              void* d_out, int out_size, void* d_ws, size_t ws_size,
                              hipStream_t stream) {
  const float* hidden = (const float*)d_in[0];
  const float* ns1    = (const float*)d_in[1];
  const float* Wq     = (const float*)d_in[2];
  const float* bq     = (const float*)d_in[3];
  const float* Wk     = (const float*)d_in[4];
  const float* bk     = (const float*)d_in[5];
  const float* Wv     = (const float*)d_in[6];
  const float* bv     = (const float*)d_in[7];
  const float* Wo     = (const float*)d_in[8];
  const float* bo     = (const float*)d_in[9];
  const float* sinks  = (const float*)d_in[10];
  const float* ns2    = (const float*)d_in[11];
  const float* rW     = (const float*)d_in[12];
  const float* rb     = (const float*)d_in[13];
  const float* gup    = (const float*)d_in[14];
  const float* gub    = (const float*)d_in[15];
  const float* down   = (const float*)d_in[16];
  const float* dbias  = (const float*)d_in[17];
  float* out = (float*)d_out;

  char* w = (char*)d_ws;
  size_t off = 0;
  auto alloc = [&](size_t bytes) -> void* {
    void* p = w + off;
    off += (bytes + 255) & ~(size_t)255;
    return p;
  };
  __bf16* xn    = (__bf16*)alloc((size_t)TNUM * HID * 2);
  float*  qkv   = (float*) alloc((size_t)TNUM * 1536 * 4);
  __bf16* qbuf  = (__bf16*)alloc((size_t)TNUM * HID * 2);
  __bf16* kbuf  = (__bf16*)alloc((size_t)TNUM * 256 * 2);
  __bf16* vbuf  = (__bf16*)alloc((size_t)TNUM * 256 * 2);
  __bf16* attnb = (__bf16*)alloc((size_t)TNUM * HID * 2);  // reused as "gated" in MoE
  float*  scr   = (float*) alloc((size_t)TNUM * 8 * 4);

  // 1) rmsnorm1
  rmsnorm_kernel<<<TNUM, 256, 0, stream>>>(hidden, ns1, xn);
  // 2) QKV projections (bias in epilogue), fp32 qkv buffer [T,1536]
  gemm_kernel<<<dim3(32, 8), 256, 0, stream>>>(xn, HID, Wq, 1024, bq, qkv, 1536, 0,
                                               nullptr, nullptr, 0, HID, 0);
  gemm_kernel<<<dim3(32, 2), 256, 0, stream>>>(xn, HID, Wk, 256, bk, qkv, 1536, 1024,
                                               nullptr, nullptr, 0, HID, 0);
  gemm_kernel<<<dim3(32, 2), 256, 0, stream>>>(xn, HID, Wv, 256, bv, qkv, 1536, 1280,
                                               nullptr, nullptr, 0, HID, 0);
  // 3) RoPE * scale -> bf16 q/k/v
  rope_kernel<<<TNUM, 256, 0, stream>>>(qkv, qbuf, kbuf, vbuf);
  // 4) attention (sliding window + sink)
  attn_kernel<<<dim3(SEQ / 16, 16, 2), 32, 0, stream>>>(qbuf, kbuf, vbuf, sinks, attnb);
  // 5) Wo projection + residual -> d_out
  gemm_kernel<<<dim3(32, 8), 256, 0, stream>>>(attnb, HID, Wo, 1024, bo, out, HID, 0,
                                               hidden, nullptr, 0, HID, 1);
  // 6) rmsnorm2 -> xn (bf16 for MoE GEMMs)
  rmsnorm_kernel<<<TNUM, 256, 0, stream>>>(out, ns2, xn);
  // 7) router (fp32 rmsnorm recomputed internally)
  router_kernel<<<TNUM, 256, 0, stream>>>(out, ns2, rW, rb, scr);
  // 8) experts: gate_up+GLU -> gated, then down GEMM accumulating score-weighted into d_out
  for (int e = 0; e < 8; ++e) {
    gateup_kernel<<<dim3(32, 16), 256, 0, stream>>>(
        xn, gup + (size_t)e * 1024 * 2048, gub + (size_t)e * 2048, attnb);
    gemm_kernel<<<dim3(32, 8), 256, 0, stream>>>(
        attnb, HID, down + (size_t)e * 1024 * 1024, 1024, dbias + (size_t)e * 1024,
        out, HID, 0, nullptr, scr, e, HID, 2);
  }
  (void)in_sizes; (void)n_in; (void)out_size; (void)ws_size;
}
